// MPCActor_60559038873600
// MI455X (gfx1250) — matrix operand order repaired and checked by reading, not compiled, not run on hardware
//
#include <hip/hip_runtime.h>
#include <math.h>

typedef __attribute__((ext_vector_type(16))) _Float16 v16h;
typedef __attribute__((ext_vector_type(8)))  _Float16 v8h;
typedef __attribute__((ext_vector_type(8)))  float    v8f;

#define BATCH 65536
#define OBSD  22
#define HID   512
#define SDIM  12
#define CDIM  4
#define NDIM  16
#define TSTEPS 5

// ---------------- WMMA fragment loaders (CDNA5 16x16x32 f16 layouts) -------

// A (16x32 slice): lanes 0-15 -> M=lane, K base 0 ; lanes 16-31 -> M=lane-16, K base 8
// per-lane halves: elems 0..7 = K kb..kb+7, elems 8..15 = K kb+16..kb+23
static __device__ inline v16h load_a_frag(const _Float16* __restrict__ base, int ld, int lane) {
  const int r  = lane & 15;
  const int kb = (lane >> 4) << 3;            // 0 or 8
  const _Float16* p = base + (size_t)r * ld + kb;
  v8h lo = *(const v8h*)(p);
  v8h hi = *(const v8h*)(p + 16);
  v16h a;
#pragma unroll
  for (int i = 0; i < 8; ++i) { a[i] = lo[i]; a[i + 8] = hi[i]; }
  return a;
}

// B (32x16 tile) from WT stored [N][K] row-major: lane holds column n = lane&15,
// elems 0..7 = K kb..kb+7, elems 8..15 = K kb+16..kb+23, kb = 0 (lanes 0-15) / 8 (lanes 16-31)
static __device__ inline v16h load_b_frag(const _Float16* __restrict__ baseNT, int ld, int lane) {
  const int n  = lane & 15;
  const int kb = (lane >> 4) << 3;            // 0 or 8
  const _Float16* p = baseNT + (size_t)n * ld + kb;
  v8h lo = *(const v8h*)(p);
  v8h hi = *(const v8h*)(p + 16);
  v16h b;
#pragma unroll
  for (int i = 0; i < 8; ++i) { b[i] = lo[i]; b[i + 8] = hi[i]; }
  return b;
}

// ---------------- conversion / transpose kernels ---------------------------

__global__ void cvt_obs(const float* __restrict__ obs, _Float16* __restrict__ obsH) {
  int i = blockIdx.x * blockDim.x + threadIdx.x;      // over BATCH*32
  int b = i >> 5, k = i & 31;
  obsH[i] = (k < OBSD) ? (_Float16)obs[(size_t)b * OBSD + k] : (_Float16)0.0f;
}
__global__ void cvt_w1(const float* __restrict__ W1, _Float16* __restrict__ W1T) {
  int i = blockIdx.x * blockDim.x + threadIdx.x;      // over HID*32 ; W1T[n][k]
  int n = i >> 5, k = i & 31;
  W1T[i] = (k < OBSD) ? (_Float16)W1[(size_t)k * HID + n] : (_Float16)0.0f;
}
__global__ void cvt_w2(const float* __restrict__ W2, _Float16* __restrict__ W2T) {
  int i = blockIdx.x * blockDim.x + threadIdx.x;      // over HID*HID ; W2T[n][k]
  int n = i >> 9, k = i & 511;
  W2T[i] = (_Float16)W2[(size_t)k * HID + n];
}
__global__ void cvt_w3(const float* __restrict__ W3, _Float16* __restrict__ W3T) {
  int i = blockIdx.x * blockDim.x + threadIdx.x;      // over 32*HID ; W3T[n][k]
  int n = i >> 9, k = i & 511;
  W3T[i] = (_Float16)W3[(size_t)k * 32 + n];
}

// ---------------- GEMM + bias + ReLU (f16 out), generic K ------------------
// Wave computes a 64(M)x64(N) tile: 4x4 grid of 16x16 WMMA accumulators.
// Each B fragment feeds 4 WMMAs (4x less B traffic than a 16-row tile);
// the 4 waves of a block share the same 64 A rows (different 64-col groups),
// so A fragment reads hit in WGP$. Block tile = 64x256, grid.y covers 512 cols.
__global__ __launch_bounds__(128) void gemm_h_relu(
    const _Float16* __restrict__ Ain, const _Float16* __restrict__ WT,
    const float* __restrict__ bias, _Float16* __restrict__ Out, int Kdim)
{
  const int lane = threadIdx.x & 31;
  const int wave = threadIdx.x >> 5;
  const int row0 = blockIdx.x << 6;                       // 64 rows per block
  const int colBase = (blockIdx.y << 8) + (wave << 6);    // 64 cols per wave

  v8f acc[4][4] = {};
  for (int k0 = 0; k0 < Kdim; k0 += 32) {
    v16h a[4];
#pragma unroll
    for (int i = 0; i < 4; ++i)
      a[i] = load_a_frag(Ain + (size_t)(row0 + i * 16) * Kdim + k0, Kdim, lane);
#pragma unroll
    for (int j = 0; j < 4; ++j) {
      v16h b = load_b_frag(WT + (size_t)(colBase + j * 16) * Kdim + k0, Kdim, lane);
#pragma unroll
      for (int i = 0; i < 4; ++i)
        acc[i][j] = __builtin_amdgcn_wmma_f32_16x16x32_f16(
            false, a[i], false, b, (short)0, acc[i][j], false, false);
    }
  }
  const int nloc = lane & 15;
  const int moff = (lane >> 4) << 3;
#pragma unroll
  for (int j = 0; j < 4; ++j) {
    const int col = colBase + j * 16 + nloc;
    const float bb = bias[col];
#pragma unroll
    for (int i = 0; i < 4; ++i) {
#pragma unroll
      for (int r = 0; r < 8; ++r) {
        float vv = acc[i][j][r] + bb;
        vv = vv > 0.0f ? vv : 0.0f;
        Out[(size_t)(row0 + i * 16 + moff + r) * HID + col] = (_Float16)vv;
      }
    }
  }
}

// ---------------- GEMM3 + bias + sigmoid (f32 out, N=32) -------------------
// Wave computes 64(M)x32(N): 4x2 accumulators; block covers 256 rows.
__global__ __launch_bounds__(128) void gemm_sigmoid(
    const _Float16* __restrict__ Ain, const _Float16* __restrict__ WT,
    const float* __restrict__ bias, float* __restrict__ Out)
{
  const int lane = threadIdx.x & 31;
  const int wave = threadIdx.x >> 5;
  const int row0 = ((blockIdx.x << 2) + wave) << 6;       // 64 rows per wave

  v8f acc[4][2] = {};
  for (int k0 = 0; k0 < HID; k0 += 32) {
    v16h a[4];
#pragma unroll
    for (int i = 0; i < 4; ++i)
      a[i] = load_a_frag(Ain + (size_t)(row0 + i * 16) * HID + k0, HID, lane);
#pragma unroll
    for (int j = 0; j < 2; ++j) {
      v16h b = load_b_frag(WT + (size_t)(j * 16) * HID + k0, HID, lane);
#pragma unroll
      for (int i = 0; i < 4; ++i)
        acc[i][j] = __builtin_amdgcn_wmma_f32_16x16x32_f16(
            false, a[i], false, b, (short)0, acc[i][j], false, false);
    }
  }
  const int nloc = lane & 15;
  const int moff = (lane >> 4) << 3;
#pragma unroll
  for (int j = 0; j < 2; ++j) {
    const int col = j * 16 + nloc;
    const float bb = bias[col];
#pragma unroll
    for (int i = 0; i < 4; ++i) {
#pragma unroll
      for (int r = 0; r < 8; ++r) {
        float vv = acc[i][j][r] + bb;
        vv = 1.0f / (1.0f + __expf(-vv));
        Out[(size_t)(row0 + i * 16 + moff + r) * 32 + col] = vv;
      }
    }
  }
}

// ---------------- per-batch LQR (one thread per batch element) -------------
// Uses the exact-arithmetic Riccati identity: with K = -Quu^{-1} Qux,
// V_new = Qxx + Qxu K and v_new = qx + Qxu k (Qt symmetric throughout).
__global__ __launch_bounds__(128) void mpc_kernel(
    const float* __restrict__ y32, const float* __restrict__ x_init,
    const float* __restrict__ Amat, const float* __restrict__ Bmat,
    float* __restrict__ uout)
{
  __shared__ float F[SDIM][NDIM];   // F = [A | Bm], 12x16
  for (int i = threadIdx.x; i < SDIM * NDIM; i += blockDim.x) {
    int s = i / NDIM, m = i % NDIM;
    F[s][m] = (m < SDIM) ? Amat[s * SDIM + m] : Bmat[s * CDIM + (m - SDIM)];
  }
  __syncthreads();

  const int b = blockIdx.x * blockDim.x + threadIdx.x;

  float q[NDIM], p[NDIM];
  for (int i = 0; i < NDIM; ++i) {
    q[i] = y32[(size_t)b * 32 + i];
    p[i] = y32[(size_t)b * 32 + NDIM + i];
  }

  float V[SDIM][SDIM], v[SDIM];
  for (int i = 0; i < SDIM; ++i) { v[i] = 0.0f; for (int j = 0; j < SDIM; ++j) V[i][j] = 0.0f; }

  float Kall[TSTEPS][CDIM][SDIM];
  float kall[TSTEPS][CDIM];

  for (int it = 0; it < TSTEPS; ++it) {
    // M = V @ F (12x16)
    float M[SDIM][NDIM];
    for (int s = 0; s < SDIM; ++s)
      for (int n = 0; n < NDIM; ++n) {
        float acc = 0.0f;
        for (int t = 0; t < SDIM; ++t) acc += V[s][t] * F[t][n];
        M[s][n] = acc;
      }
    // Qt = diag(q) + F^T M (16x16),  qt = p + F^T v
    float Qt[NDIM][NDIM], qt[NDIM];
    for (int m = 0; m < NDIM; ++m) {
      for (int n = 0; n < NDIM; ++n) {
        float acc = (m == n) ? q[m] : 0.0f;
        for (int s = 0; s < SDIM; ++s) acc += F[s][m] * M[s][n];
        Qt[m][n] = acc;
      }
      float accq = p[m];
      for (int s = 0; s < SDIM; ++s) accq += F[s][m] * v[s];
      qt[m] = accq;
    }
    // Solve Quu * X = [Qux | qu]   (Quu SPD 4x4; Gaussian elim, no pivot)
    float Q[CDIM][CDIM], R[CDIM][SDIM + 1];
    for (int c = 0; c < CDIM; ++c) {
      for (int d = 0; d < CDIM; ++d) Q[c][d] = Qt[SDIM + c][SDIM + d];
      for (int s = 0; s < SDIM; ++s) R[c][s] = Qt[SDIM + c][s];
      R[c][SDIM] = qt[SDIM + c];
    }
    for (int i = 0; i < CDIM; ++i) {
      float inv = 1.0f / Q[i][i];
      for (int j = i + 1; j < CDIM; ++j) {
        float f = Q[j][i] * inv;
        for (int d = i; d < CDIM; ++d) Q[j][d] -= f * Q[i][d];
        for (int d = 0; d <= SDIM; ++d) R[j][d] -= f * R[i][d];
      }
    }
    float X[CDIM][SDIM + 1];
    for (int i = CDIM - 1; i >= 0; --i) {
      float inv = 1.0f / Q[i][i];
      for (int d = 0; d <= SDIM; ++d) {
        float acc = R[i][d];
        for (int j = i + 1; j < CDIM; ++j) acc -= Q[i][j] * X[j][d];
        X[i][d] = acc * inv;
      }
    }
    for (int c = 0; c < CDIM; ++c) {
      for (int s = 0; s < SDIM; ++s) Kall[it][c][s] = -X[c][s];
      kall[it][c] = -X[c][SDIM];
    }
    // V_new = Qxx + Qxu K ; v_new = qx + Qxu k
    float Vn[SDIM][SDIM], vn[SDIM];
    for (int s = 0; s < SDIM; ++s) {
      for (int m = 0; m < SDIM; ++m) {
        float acc = Qt[s][m];
        for (int c = 0; c < CDIM; ++c) acc += Qt[s][SDIM + c] * Kall[it][c][m];
        Vn[s][m] = acc;
      }
      float acc = qt[s];
      for (int c = 0; c < CDIM; ++c) acc += Qt[s][SDIM + c] * kall[it][c];
      vn[s] = acc;
    }
    for (int s = 0; s < SDIM; ++s) { v[s] = vn[s]; for (int m = 0; m < SDIM; ++m) V[s][m] = Vn[s][m]; }
  }

  // forward rollout (gains consumed in reversed order)
  float x[SDIM];
  for (int s = 0; s < SDIM; ++s) x[s] = x_init[(size_t)b * SDIM + s];
  for (int t = 0; t < TSTEPS; ++t) {
    const int it = TSTEPS - 1 - t;
    float u[CDIM];
    for (int c = 0; c < CDIM; ++c) {
      float acc = kall[it][c];
      for (int s = 0; s < SDIM; ++s) acc += Kall[it][c][s] * x[s];
      u[c] = acc;
      uout[((size_t)t * BATCH + b) * CDIM + c] = acc;
    }
    float xn[SDIM];
    for (int s = 0; s < SDIM; ++s) {
      float acc = 0.0f;
      for (int j = 0; j < SDIM; ++j) acc += F[s][j] * x[j];
      for (int c = 0; c < CDIM; ++c) acc += F[s][SDIM + c] * u[c];
      xn[s] = acc;
    }
    for (int s = 0; s < SDIM; ++s) x[s] = xn[s];
  }
}

// ---------------- host launcher --------------------------------------------

extern "C" void kernel_launch(void* const* d_in, const int* in_sizes, int n_in,
                              void* d_out, int out_size, void* d_ws, size_t ws_size,
                              hipStream_t stream)
{
  (void)in_sizes; (void)n_in; (void)out_size; (void)ws_size;

  const float* obs    = (const float*)d_in[0];
  const float* x_init = (const float*)d_in[1];
  const float* W1     = (const float*)d_in[2];
  const float* b1     = (const float*)d_in[3];
  const float* W2     = (const float*)d_in[4];
  const float* b2     = (const float*)d_in[5];
  const float* W3     = (const float*)d_in[6];
  const float* b3     = (const float*)d_in[7];
  const float* Amat   = (const float*)d_in[8];
  const float* Bmat   = (const float*)d_in[9];
  float* uout = (float*)d_out;

  char* ws = (char*)d_ws;
  size_t off = 0;
  auto take = [&](size_t bytes) -> void* {
    void* ptr = ws + off;
    off += (bytes + 255) & ~(size_t)255;
    return ptr;
  };
  _Float16* obsH = (_Float16*)take((size_t)BATCH * 32  * sizeof(_Float16));
  _Float16* W1T  = (_Float16*)take((size_t)HID   * 32  * sizeof(_Float16));
  _Float16* W2T  = (_Float16*)take((size_t)HID   * HID * sizeof(_Float16));
  _Float16* W3T  = (_Float16*)take((size_t)32    * HID * sizeof(_Float16));
  _Float16* act1 = (_Float16*)take((size_t)BATCH * HID * sizeof(_Float16));
  _Float16* act2 = (_Float16*)take((size_t)BATCH * HID * sizeof(_Float16));
  float*    y32  = (float*)   take((size_t)BATCH * 32  * sizeof(float));

  cvt_obs<<<(BATCH * 32) / 256, 256, 0, stream>>>(obs, obsH);
  cvt_w1 <<<(HID * 32)   / 256, 256, 0, stream>>>(W1, W1T);
  cvt_w2 <<<(HID * HID)  / 256, 256, 0, stream>>>(W2, W2T);
  cvt_w3 <<<(HID * 32)   / 256, 256, 0, stream>>>(W3, W3T);

  // Layer 1: K padded 22->32 (single K-step, same kernel)
  gemm_h_relu<<<dim3(BATCH / 64, 2), 128, 0, stream>>>(obsH, W1T, b1, act1, 32);
  // Layer 2: the dominant 34 GFLOP GEMM, 64x64 tile per wave
  gemm_h_relu<<<dim3(BATCH / 64, 2), 128, 0, stream>>>(act1, W2T, b2, act2, HID);
  // Layer 3: N=32, sigmoid, f32 out, 64x32 tile per wave
  gemm_sigmoid<<<dim3(BATCH / 256), 128, 0, stream>>>(act2, W3T, b3, y32);
  // Batched LQR
  mpc_kernel<<<BATCH / 128, 128, 0, stream>>>(y32, x_init, Amat, Bmat, uout);
}
